// SingleHead_20289425507037
// MI455X (gfx1250) — compile-verified, matrix operand order checked by reading, not run on hardware
//
#include <hip/hip_runtime.h>

typedef __attribute__((ext_vector_type(16))) _Float16 v16h;
typedef __attribute__((ext_vector_type(8)))  _Float16 v8h;
typedef __attribute__((ext_vector_type(8)))  float    v8f;
typedef __attribute__((ext_vector_type(4)))  int      v4i;

#define B_ 4
#define S_ 4096
#define D_ 1024
#define H_ 128

#if __has_builtin(__builtin_amdgcn_global_load_async_to_lds_b128)
#define HAVE_ASYNC_LDS 1
#else
#define HAVE_ASYNC_LDS 0
#endif

#if HAVE_ASYNC_LDS
#if __has_builtin(__builtin_amdgcn_s_wait_asynccnt)
#define WAIT_ASYNC(n) __builtin_amdgcn_s_wait_asynccnt(n)
#else
#define WAIT_ASYNC(n) asm volatile("s_wait_asynccnt " #n ::: "memory")
#endif
#else
#define WAIT_ASYNC(n) ((void)0)
#endif

// Copy one 16-byte chunk global -> LDS (async DMA when available).
static __device__ __forceinline__ void stage16(const _Float16* g, _Float16* l) {
#if HAVE_ASYNC_LDS
    __builtin_amdgcn_global_load_async_to_lds_b128(
        (__attribute__((address_space(1))) v4i*)g,
        (__attribute__((address_space(3))) v4i*)l, 0, 0);
#else
    *reinterpret_cast<v8h*>(l) = *reinterpret_cast<const v8h*>(g);
#endif
}

static __device__ __forceinline__ v8f wmma_f16(v16h a, v16h b, v8f c) {
    // (neg_a, A, neg_b, B, c_mod, C, reuse_a, reuse_b)
    return __builtin_amdgcn_wmma_f32_16x16x32_f16(false, a, false, b, (short)0, c,
                                                  false, false);
}

// ---------------------------------------------------------------------------
// Kernel 0: weight prep. Wq/Wk/Wv [D,H] f32 -> Wt[mat][H][D] f16 (transposed)
// so projection B-fragments become contiguous 16-byte loads.
// ---------------------------------------------------------------------------
__global__ __launch_bounds__(256) void wt_prep_kernel(
    const float* __restrict__ Wq, const float* __restrict__ Wk,
    const float* __restrict__ Wv, _Float16* __restrict__ Wt)
{
    const int idx = blockIdx.x * 256 + threadIdx.x;   // [0, 3*D*H)
    const int m   = idx / (D_ * H_);
    const int r   = idx % (D_ * H_);
    const int d   = r / H_;
    const int h   = r % H_;
    const float* src = (m == 0) ? Wq : (m == 1) ? Wk : Wv;
    Wt[(size_t)m * H_ * D_ + (size_t)h * D_ + d] = (_Float16)src[(size_t)d * H_ + h];
}

// ---------------------------------------------------------------------------
// Kernel 1: QKV projection. One wave computes a full 16x128 slab (8 col-tiles)
// of one matrix for one 16-row tile: 8 chained WMMAs per K-step, A-fragment
// amortized across all 8. q,k stored row-major f16; v stored transposed
// (vt[batch][H][S]) for contiguous PV B-fragment loads in the flash kernel.
// ---------------------------------------------------------------------------
__global__ __launch_bounds__(256) void qkv_proj_kernel(
    const float* __restrict__ x,
    const _Float16* __restrict__ Wt,
    const float* __restrict__ bq, const float* __restrict__ bk,
    const float* __restrict__ bv,
    _Float16* __restrict__ qh, _Float16* __restrict__ kh,
    _Float16* __restrict__ vt)
{
    const int lane = threadIdx.x & 31;
    const int half = lane >> 4;
    const int m16  = lane & 15;

    const int w        = blockIdx.x * 8 + (threadIdx.x >> 5);  // [0, 3072)
    const int row_tile = w / 3;    // [0, 1024)
    const int mat      = w % 3;    // 0=q 1=k 2=v

    const _Float16* Wm = Wt + (size_t)mat * H_ * D_;
    const float*    bm = (mat == 0) ? bq : (mat == 1) ? bk : bv;

    const int    row = row_tile * 16 + m16;   // A row handled by this lane
    const float* xr  = x + (size_t)row * D_;

    v8f acc[8];
    #pragma unroll
    for (int h = 0; h < 8; ++h) acc[h] = (v8f){};

    for (int kk = 0; kk < D_; kk += 32) {
        // A fragment: 16x32 f16 (ISA layout: lanes 0-15 K 0-7/16-23,
        // lanes 16-31 K 8-15/24-31), converted from f32 x.
        v16h a;
        const float4* pa0 = reinterpret_cast<const float4*>(xr + kk + half * 8);
        const float4* pa1 = reinterpret_cast<const float4*>(xr + kk + 16 + half * 8);
        float4 f0 = pa0[0], f1 = pa0[1];
        float4 f2 = pa1[0], f3 = pa1[1];
        a[0]=(_Float16)f0.x;  a[1]=(_Float16)f0.y;  a[2]=(_Float16)f0.z;  a[3]=(_Float16)f0.w;
        a[4]=(_Float16)f1.x;  a[5]=(_Float16)f1.y;  a[6]=(_Float16)f1.z;  a[7]=(_Float16)f1.w;
        a[8]=(_Float16)f2.x;  a[9]=(_Float16)f2.y;  a[10]=(_Float16)f2.z; a[11]=(_Float16)f2.w;
        a[12]=(_Float16)f3.x; a[13]=(_Float16)f3.y; a[14]=(_Float16)f3.z; a[15]=(_Float16)f3.w;

        #pragma unroll
        for (int h = 0; h < 8; ++h) {
            // B fragment from transposed weights: contiguous b128 loads.
            const _Float16* wrow = Wm + (size_t)(h * 16 + m16) * D_ + kk;
            v8h lo = *reinterpret_cast<const v8h*>(wrow + half * 8);
            v8h hi = *reinterpret_cast<const v8h*>(wrow + 16 + half * 8);
            v16h b;
            #pragma unroll
            for (int e = 0; e < 8; ++e) { b[e] = lo[e]; b[8 + e] = hi[e]; }
            acc[h] = wmma_f16(a, b, acc[h]);
        }
    }

    #pragma unroll
    for (int h = 0; h < 8; ++h) {
        const int   col  = h * 16 + m16;
        const float bias = bm[col];
        #pragma unroll
        for (int r = 0; r < 8; ++r) {
            const int grow = row_tile * 16 + r + 8 * half;  // C layout: M = r + 8*half
            const _Float16 val = (_Float16)(acc[h][r] + bias);
            if (mat == 0) {
                qh[(size_t)grow * H_ + col] = val;
            } else if (mat == 1) {
                kh[(size_t)grow * H_ + col] = val;
            } else {
                const int batch = grow >> 12;        // / S_
                const int s     = grow & (S_ - 1);
                vt[(size_t)batch * H_ * S_ + (size_t)col * S_ + s] = val;
            }
        }
    }
}

// ---------------------------------------------------------------------------
// Kernel 2: causal flash attention, cooperative form.
// A block (4 waves) owns a 64-query slab; K/V blocks of 32 keys are staged
// into double-buffered LDS with async global->LDS DMA (ASYNCcnt pipeline)
// shared by all waves. Each wave computes a 16-query tile: QK^T and PV via
// WMMA f16->f32 with all B-fragments read from LDS; online softmax with
// 16-lane butterfly reductions; P tile re-laid-out C->A through LDS.
// ---------------------------------------------------------------------------
__global__ __launch_bounds__(128) void flash_attn_kernel(
    const _Float16* __restrict__ qh, const _Float16* __restrict__ kh,
    const _Float16* __restrict__ vt, float* __restrict__ out)
{
    __shared__ _Float16 kbuf[2][32 * 128];  // [buf][key][d]     8 KB each
    __shared__ _Float16 vbuf[2][128 * 32];  // [buf][h][key]     8 KB each
    __shared__ _Float16 pbuf[4][16 * 32];   // per-wave P staging

    const int tid  = threadIdx.x;
    const int lane = tid & 31;
    const int half = lane >> 4;
    const int m16  = lane & 15;
    const int wv   = tid >> 5;

    const int batch = blockIdx.x >> 6;        // 64 q-rows per block
    const int qblk  = blockIdx.x & 63;
    const int qrow0 = qblk * 64 + wv * 16;    // this wave's 16-query tile

    const _Float16* qp  = qh + (size_t)batch * S_ * H_;
    const _Float16* kp  = kh + (size_t)batch * S_ * H_;
    const _Float16* vtp = vt + (size_t)batch * H_ * S_;
    float*          op  = out + (size_t)batch * S_ * H_;

    // Q fragments (16x128 as 4 chunks of 16x32), pre-scaled by 1/sqrt(H).
    const float qscale = 0.088388347648318447f;   // 1/sqrt(128)
    const int   qrow   = qrow0 + m16;
    v16h qf[4];
    #pragma unroll
    for (int c = 0; c < 4; ++c) {
        const _Float16* qr = qp + (size_t)qrow * H_ + c * 32;
        v8h lo = *reinterpret_cast<const v8h*>(qr + half * 8);
        v8h hi = *reinterpret_cast<const v8h*>(qr + 16 + half * 8);
        #pragma unroll
        for (int e = 0; e < 8; ++e) {
            qf[c][e]     = (_Float16)((float)lo[e] * qscale);
            qf[c][8 + e] = (_Float16)((float)hi[e] * qscale);
        }
    }

    v8f o[8];
    #pragma unroll
    for (int h = 0; h < 8; ++h) o[h] = (v8f){};
    float mrow[8], lrow[8];
    #pragma unroll
    for (int r = 0; r < 8; ++r) { mrow[r] = -__builtin_inff(); lrow[r] = 0.0f; }

    // Block-uniform causal limit (covers the slab's last row).
    const int kb_max = (qblk * 64 + 63) >> 5;

    // --- staging: 8 chunks of 16B per thread per 32-key block (K:4, V:4) ---
    auto stage_block = [&](int key0, int nb) {
        #pragma unroll
        for (int i = 0; i < 4; ++i) {
            const int c = tid + i * 128;                  // [0,512): K chunk
            stage16(kp + (size_t)key0 * H_ + c * 8, &kbuf[nb][c * 8]);
        }
        #pragma unroll
        for (int i = 0; i < 4; ++i) {
            const int c   = tid + i * 128;                // [0,512): V chunk
            const int hh  = c >> 2;
            const int off = (c & 3) * 8;
            stage16(vtp + (size_t)hh * S_ + key0 + off, &vbuf[nb][hh * 32 + off]);
        }
    };

    // Prologue: stage block 0.
    stage_block(0, 0);
    WAIT_ASYNC(0);
    __syncthreads();

    for (int kb = 0; kb <= kb_max; ++kb) {
        const int key0 = kb * 32;
        const int cur  = kb & 1;

        __syncthreads();   // everyone done reading buf[cur^1] from last iter
        if (kb < kb_max) {
            stage_block(key0 + 32, cur ^ 1);
            WAIT_ASYNC(8);  // retire the 8 in-order issues for block kb
        } else {
            WAIT_ASYNC(0);
        }
        __syncthreads();   // block kb fully resident in buf[cur]

        // ---- S = Q K^T : two 16x16 score tiles over H=128 (4 k-steps) ----
        v8f s0 = {}, s1 = {};
        #pragma unroll
        for (int c = 0; c < 4; ++c) {
            const _Float16* kr0 = &kbuf[cur][(m16)      * 128 + c * 32];
            const _Float16* kr1 = &kbuf[cur][(16 + m16) * 128 + c * 32];
            v8h a0 = *reinterpret_cast<const v8h*>(kr0 + half * 8);
            v8h a1 = *reinterpret_cast<const v8h*>(kr0 + 16 + half * 8);
            v8h c0 = *reinterpret_cast<const v8h*>(kr1 + half * 8);
            v8h c1 = *reinterpret_cast<const v8h*>(kr1 + 16 + half * 8);
            v16h b0, b1;
            #pragma unroll
            for (int e = 0; e < 8; ++e) {
                b0[e] = a0[e]; b0[8 + e] = a1[e];
                b1[e] = c0[e]; b1[8 + e] = c1[e];
            }
            s0 = wmma_f16(qf[c], b0, s0);
            s1 = wmma_f16(qf[c], b1, s1);
        }

        // ---- causal mask + online softmax (rows split across lane halves) ----
        #pragma unroll
        for (int r = 0; r < 8; ++r) {
            const int   qr2 = qrow0 + r + 8 * half;      // C layout: M = r + 8*half
            const float x0 = (key0 + m16      <= qr2) ? s0[r] : -__builtin_inff();
            const float x1 = (key0 + 16 + m16 <= qr2) ? s1[r] : -__builtin_inff();

            float mx = fmaxf(x0, x1);
            #pragma unroll
            for (int d = 1; d < 16; d <<= 1) mx = fmaxf(mx, __shfl_xor(mx, d, 32));

            const float mn    = fmaxf(mrow[r], mx);
            const float alpha = __expf(mrow[r] - mn);
            mrow[r] = mn;

            const float p0 = __expf(x0 - mn);
            const float p1 = __expf(x1 - mn);
            float rs = p0 + p1;
            #pragma unroll
            for (int d = 1; d < 16; d <<= 1) rs += __shfl_xor(rs, d, 32);
            lrow[r] = lrow[r] * alpha + rs;

            #pragma unroll
            for (int h = 0; h < 8; ++h) o[h][r] *= alpha;

            pbuf[wv][(r + 8 * half) * 32 + m16]      = (_Float16)p0;
            pbuf[wv][(r + 8 * half) * 32 + 16 + m16] = (_Float16)p1;
        }

        // ---- reload P (16x32) as an A fragment via LDS (ds_load_b128) ----
        v16h pf;
        {
            const _Float16* bp = &pbuf[wv][m16 * 32];
            v8h lo = *reinterpret_cast<const v8h*>(bp + half * 8);
            v8h hi = *reinterpret_cast<const v8h*>(bp + 16 + half * 8);
            #pragma unroll
            for (int e = 0; e < 8; ++e) { pf[e] = lo[e]; pf[8 + e] = hi[e]; }
        }

        // ---- O += P V : 8 output column tiles, B-fragments from LDS ----
        #pragma unroll
        for (int h = 0; h < 8; ++h) {
            const _Float16* vr = &vbuf[cur][(h * 16 + m16) * 32];
            v8h lo = *reinterpret_cast<const v8h*>(vr + half * 8);
            v8h hi = *reinterpret_cast<const v8h*>(vr + 16 + half * 8);
            v16h vb;
            #pragma unroll
            for (int e = 0; e < 8; ++e) { vb[e] = lo[e]; vb[8 + e] = hi[e]; }
            o[h] = wmma_f16(pf, vb, o[h]);
        }
    }

    // ---- epilogue: normalize by running sum, write f32 output ----
    #pragma unroll
    for (int r = 0; r < 8; ++r) {
        const float inv  = 1.0f / lrow[r];
        const int   orow = qrow0 + r + 8 * half;
        #pragma unroll
        for (int h = 0; h < 8; ++h)
            op[(size_t)orow * H_ + h * 16 + m16] = o[h][r] * inv;
    }
}

extern "C" void kernel_launch(void* const* d_in, const int* in_sizes, int n_in,
                              void* d_out, int out_size, void* d_ws, size_t ws_size,
                              hipStream_t stream) {
    const float* x  = (const float*)d_in[0];
    const float* Wq = (const float*)d_in[1];
    const float* bq = (const float*)d_in[2];
    const float* Wk = (const float*)d_in[3];
    const float* bk = (const float*)d_in[4];
    const float* Wv = (const float*)d_in[5];
    const float* bv = (const float*)d_in[6];
    float* out = (float*)d_out;

    // Workspace layout (f16): q, k row-major; v transposed; weights transposed.
    const size_t nqkv = (size_t)B_ * S_ * H_;     // 2,097,152 elems each
    _Float16* qh = (_Float16*)d_ws;
    _Float16* kh = qh + nqkv;
    _Float16* vt = kh + nqkv;                     // [B][H][S]
    _Float16* Wt = vt + nqkv;                     // [3][H][D]

    // Kernel 0: weight transpose+convert (3*D*H = 393216 elements).
    wt_prep_kernel<<<(3 * D_ * H_) / 256, 256, 0, stream>>>(Wq, Wk, Wv, Wt);

    // Kernel 1: 1024 row-tiles * 3 mats = 3072 waves, 8 waves/block.
    qkv_proj_kernel<<<3072 / 8, 256, 0, stream>>>(x, Wt, bq, bk, bv, qh, kh, vt);

    // Kernel 2: 4 waves per 128-thread block, one 64-query slab per block.
    flash_attn_kernel<<<B_ * (S_ / 64), 128, 0, stream>>>(qh, kh, vt, out);
}
